// RouteNet_temporal_delay_2156073582595
// MI455X (gfx1250) — matrix operand order repaired and checked by reading, not compiled, not run on hardware
//
#include <hip/hip_runtime.h>
#include <hip/hip_bf16.h>

#define NFLOW 16384
#define NLINK 2048
#define LPATH 8
#define SEGS  3
#define ITERS 8
#define DDIM  32
#define FPL   64

typedef _Float16 v8h  __attribute__((ext_vector_type(8)));
typedef _Float16 v16h __attribute__((ext_vector_type(16)));
typedef float    v8f  __attribute__((ext_vector_type(8)));

__device__ __forceinline__ float sigmoidf_(float x) { return 1.0f / (1.0f + __expf(-x)); }
__device__ __forceinline__ float tanhf_(float x)    { return 2.0f * sigmoidf_(2.0f * x) - 1.0f; }
__device__ __forceinline__ float reluf_(float x)    { return fmaxf(x, 0.0f); }

__device__ __forceinline__ v16h cat8(v8h lo, v8h hi) {
  v16h r;
#pragma unroll
  for (int i = 0; i < 8; ++i) { r[i] = lo[i]; r[i + 8] = hi[i]; }
  return r;
}

// A-fragment (16 x 32 f16, K-chunk starting at kc) from LDS tile [16][pitch] (row-major).
// ISA 16-bit A layout: lanes 0-15 hold K kb..kb+7 (v0-3) and kb+16..kb+23 (v4-7), kb=kc;
// lanes 16-31 take kb = kc+8.
__device__ __forceinline__ v16h ldA(const _Float16* base, int pitch, int lane, int kc) {
  const int row = lane & 15;
  const int kb  = kc + ((lane >> 4) << 3);
  const _Float16* p = base + row * pitch + kb;
  v8h lo = *reinterpret_cast<const v8h*>(p);
  v8h hi = *reinterpret_cast<const v8h*>(p + 16);
  return cat8(lo, hi);
}

// B-fragment (32 x 16 f16) for output tile ntile from LDS W^T [96][pitch] (n-major, k contiguous).
// Layout matches A: lane = N (l%16); lanes 0-15 K kc..kc+7, kc+16..kc+23; lanes 16-31 K kc+8..kc+15, kc+24..kc+31.
__device__ __forceinline__ v16h ldB(const _Float16* wT, int pitch, int lane, int ntile, int kc) {
  const int n  = ntile * 16 + (lane & 15);
  const int kb = kc + ((lane >> 4) << 3);
  const _Float16* p = wT + n * pitch + kb;
  v8h lo = *reinterpret_cast<const v8h*>(p);
  v8h hi = *reinterpret_cast<const v8h*>(p + 16);
  return cat8(lo, hi);
}

__device__ __forceinline__ v8f wmma16(v16h a, v16h b, v8f c) {
  return __builtin_amdgcn_wmma_f32_16x16x32_f16(false, a, false, b, (short)0, c, false, false);
}

// ---------------------------------------------------------------------------
// Embedding kernels (tiny, scalar)
// ---------------------------------------------------------------------------
__global__ __launch_bounds__(256) void flow_embed_kernel(
    const float* __restrict__ traffic, const float* __restrict__ pkt,
    const unsigned char* __restrict__ has,
    const float* __restrict__ w1, const float* __restrict__ b1,
    const float* __restrict__ w2, const float* __restrict__ b2,
    float* __restrict__ init_flow) {
  int i = blockIdx.x * 256 + threadIdx.x;
  if (i >= NFLOW * SEGS) return;
  int f = i / SEGS, s = i % SEGS;
  float x0 = traffic[f * SEGS + s];
  float x1 = pkt[f * SEGS + s];
  float x2 = has[f * SEGS + s] ? 1.0f : 0.0f;
  float h1[DDIM];
#pragma unroll
  for (int d = 0; d < DDIM; ++d)
    h1[d] = reluf_(b1[d] + x0 * w1[d] + x1 * w1[DDIM + d] + x2 * w1[2 * DDIM + d]);
  float* dst = init_flow + ((size_t)s * NFLOW + f) * DDIM;
  for (int d = 0; d < DDIM; ++d) {
    float a = b2[d];
    for (int j = 0; j < DDIM; ++j) a += h1[j] * w2[j * DDIM + d];
    dst[d] = reluf_(a);
  }
}

__global__ __launch_bounds__(256) void link_embed_kernel(
    const float* __restrict__ traffic, const float* __restrict__ pkt,
    const int* __restrict__ p2l, const float* __restrict__ cap,
    const float* __restrict__ hdr,
    const float* __restrict__ w1, const float* __restrict__ b1,
    const float* __restrict__ w2, const float* __restrict__ b2,
    float* __restrict__ init_link) {
  int l = blockIdx.x * 256 + threadIdx.x;
  if (l >= NLINK) return;
  float capacity = cap[l] * 1.0e9f;
  float hd = hdr[l];
  for (int s = 0; s < SEGS; ++s) {
    float ft = 0.0f, fp = 0.0f;
    for (int j = 0; j < FPL; ++j) {
      int f = p2l[(l * FPL + j) * 2];
      ft += traffic[f * SEGS + s];
      fp += pkt[f * SEGS + s];
    }
    float load = (ft + fp * hd) / capacity;
    float h1[DDIM];
#pragma unroll
    for (int d = 0; d < DDIM; ++d) h1[d] = reluf_(b1[d] + load * w1[d]);
    float* dst = init_link + ((size_t)s * NLINK + l) * DDIM;
    for (int d = 0; d < DDIM; ++d) {
      float a = b2[d];
      for (int j = 0; j < DDIM; ++j) a += h1[j] * w2[j * DDIM + d];
      dst[d] = reluf_(a);
    }
  }
}

__global__ __launch_bounds__(256) void queue_embed_kernel(
    const int* __restrict__ btype,
    const float* __restrict__ w1, const float* __restrict__ b1,
    const float* __restrict__ w2, const float* __restrict__ b2,
    float* __restrict__ init_queue, float* __restrict__ queue_state) {
  int l = blockIdx.x * 256 + threadIdx.x;
  if (l >= NLINK) return;
  int bt = btype[l] & 1;
  float h1[DDIM];
#pragma unroll
  for (int d = 0; d < DDIM; ++d) h1[d] = reluf_(b1[d] + w1[bt * DDIM + d]);
  for (int d = 0; d < DDIM; ++d) {
    float a = b2[d];
    for (int j = 0; j < DDIM; ++j) a += h1[j] * w2[j * DDIM + d];
    a = reluf_(a);
    init_queue[l * DDIM + d] = a;
    queue_state[l * DDIM + d] = a;
  }
}

// ---------------------------------------------------------------------------
// Flow GRU scan: one kernel per message-passing iteration.
// 128 flows/block (8 waves x 16 flows). Hidden state lives in registers
// in WMMA C-layout for the whole 8-step scan; 18 v_wmma per wave per step.
// ---------------------------------------------------------------------------
__global__ __launch_bounds__(256) void flow_gru_kernel(
    const float* __restrict__ fg_wi, const float* __restrict__ fg_ui,
    const float* __restrict__ fg_bi, const float* __restrict__ fg_br,
    const int* __restrict__ link_to_path,
    const float* __restrict__ queue_state, const float* __restrict__ link_state,
    const float* __restrict__ h_src, int h_stride,
    float* __restrict__ fss) {
  __shared__ __align__(16) _Float16 wiT[96][64];       // fg_wi^T  (n-major)
  __shared__ __align__(16) _Float16 uiT[96][32];       // fg_ui^T
  __shared__ __align__(16) _Float16 xtile[8][16][64];  // per-wave x rows
  __shared__ __align__(16) _Float16 htile[8][16][32];  // per-wave h rows

  const int tid = threadIdx.x;
  for (int i = tid; i < 96 * 64; i += 256) wiT[i >> 6][i & 63] = (_Float16)fg_wi[(i & 63) * 96 + (i >> 6)];
  for (int i = tid; i < 96 * 32; i += 256) uiT[i >> 5][i & 31] = (_Float16)fg_ui[(i & 31) * 96 + (i >> 5)];

  const int wave = tid >> 5, lane = tid & 31;
  const int ncol = lane & 15, lhalf = lane >> 4;
  const int mbase = lhalf * 8;
  const int fbase = blockIdx.x * 128 + wave * 16;

  float bzr[4], bih[2], brh[2];
#pragma unroll
  for (int j = 0; j < 4; ++j) bzr[j] = fg_bi[j * 16 + ncol] + fg_br[j * 16 + ncol];
#pragma unroll
  for (int j = 0; j < 2; ++j) {
    bih[j] = fg_bi[64 + j * 16 + ncol];
    brh[j] = fg_br[64 + j * 16 + ncol];
  }

  // Load h (C-layout: row = mbase+v, col = c*16+ncol), write fss[:,0], stage f16 to LDS.
  v8f hreg[2];
#pragma unroll
  for (int c = 0; c < 2; ++c)
#pragma unroll
    for (int v = 0; v < 8; ++v) {
      int m = fbase + mbase + v;
      float hv = h_src[(size_t)m * h_stride + c * 16 + ncol];
      hreg[c][v] = hv;
      fss[(size_t)m * (LPATH + 1) * DDIM + c * 16 + ncol] = hv;
      htile[wave][mbase + v][c * 16 + ncol] = (_Float16)hv;
    }
  __syncthreads();

  for (int t = 0; t < LPATH; ++t) {
    // Gather x = [queue_state[link], link_state[link]] (2 lanes per flow).
    {
      const int f = fbase + ncol;
      const int li = link_to_path[f * LPATH + t];
      const float* src = lhalf ? (link_state + li * DDIM) : (queue_state + li * DDIM);
      _Float16* dst = &xtile[wave][ncol][lhalf * 32];
#pragma unroll
      for (int k = 0; k < 32; ++k) dst[k] = (_Float16)src[k];
    }
    __syncthreads();

    v16h ax0 = ldA(&xtile[wave][0][0], 64, lane, 0);
    v16h ax1 = ldA(&xtile[wave][0][0], 64, lane, 32);
    v16h ah  = ldA(&htile[wave][0][0], 32, lane, 0);

    v8f zr[4], ihv[2], rhv[2];
#pragma unroll
    for (int j = 0; j < 4; ++j) {
      v8f a;
#pragma unroll
      for (int v = 0; v < 8; ++v) a[v] = bzr[j];
      a = wmma16(ax0, ldB(&wiT[0][0], 64, lane, j, 0), a);
      a = wmma16(ax1, ldB(&wiT[0][0], 64, lane, j, 32), a);
      a = wmma16(ah,  ldB(&uiT[0][0], 32, lane, j, 0), a);
      zr[j] = a;
    }
#pragma unroll
    for (int j = 0; j < 2; ++j) {
      v8f a;
#pragma unroll
      for (int v = 0; v < 8; ++v) a[v] = bih[j];
      a = wmma16(ax0, ldB(&wiT[0][0], 64, lane, 4 + j, 0), a);
      a = wmma16(ax1, ldB(&wiT[0][0], 64, lane, 4 + j, 32), a);
      ihv[j] = a;
      v8f b;
#pragma unroll
      for (int v = 0; v < 8; ++v) b[v] = brh[j];
      b = wmma16(ah, ldB(&uiT[0][0], 32, lane, 4 + j, 0), b);
      rhv[j] = b;
    }
    __syncthreads();

    // Gates: C-tiles c and c+2 share (M,N) lane mapping, so r lines up with rh.
#pragma unroll
    for (int c = 0; c < 2; ++c)
#pragma unroll
      for (int v = 0; v < 8; ++v) {
        float z = sigmoidf_(zr[c][v]);
        float r = sigmoidf_(zr[2 + c][v]);
        float hh = tanhf_(ihv[c][v] + r * rhv[c][v]);
        float hn = z * hreg[c][v] + (1.0f - z) * hh;
        hreg[c][v] = hn;
        int m = fbase + mbase + v;
        fss[(size_t)m * (LPATH + 1) * DDIM + (t + 1) * DDIM + c * 16 + ncol] = hn;
        htile[wave][mbase + v][c * 16 + ncol] = (_Float16)hn;
      }
    __syncthreads();
  }
}

// ---------------------------------------------------------------------------
// flow_sum[l][d] = sum_j fss[f_idx[l,j], pos_idx[l,j], d]
// ---------------------------------------------------------------------------
__global__ void flow_sum_kernel(const float* __restrict__ fss,
                                const int* __restrict__ p2l,
                                float* __restrict__ flow_sum) {
  int l = blockIdx.x;
  int d = threadIdx.x;  // 32 threads
  float acc = 0.0f;
  for (int j = 0; j < FPL; ++j) {
    int f = p2l[(l * FPL + j) * 2 + 0];
    int p = p2l[(l * FPL + j) * 2 + 1];
    acc += fss[(size_t)f * (LPATH + 1) * DDIM + p * DDIM + d];
  }
  flow_sum[l * DDIM + d] = acc;
}

// ---------------------------------------------------------------------------
// Generic small GRU (M = 2048 rows). kx = 32 (optionally gathered x) or
// kx = 64 (x = concat(xA[row], xB[row])). h updated in place.
// ---------------------------------------------------------------------------
__global__ __launch_bounds__(256) void small_gru_kernel(
    const float* __restrict__ wi, const float* __restrict__ ui,
    const float* __restrict__ bi, const float* __restrict__ br,
    const float* __restrict__ xA, const float* __restrict__ xB,
    const int* __restrict__ idx,
    float* __restrict__ hbuf, int kx) {
  __shared__ __align__(16) _Float16 wiT[96][64];
  __shared__ __align__(16) _Float16 uiT[96][32];
  __shared__ __align__(16) _Float16 xtile[8][16][64];
  __shared__ __align__(16) _Float16 htile[8][16][32];

  const int tid = threadIdx.x;
  for (int i = tid; i < 96 * kx; i += 256) wiT[i / kx][i % kx] = (_Float16)wi[(i % kx) * 96 + (i / kx)];
  for (int i = tid; i < 96 * 32; i += 256) uiT[i >> 5][i & 31] = (_Float16)ui[(i & 31) * 96 + (i >> 5)];

  const int wave = tid >> 5, lane = tid & 31;
  const int ncol = lane & 15, lhalf = lane >> 4;
  const int mbase = lhalf * 8;
  const int rbase = blockIdx.x * 128 + wave * 16;

  float bzr[4], bih[2], brh[2];
#pragma unroll
  for (int j = 0; j < 4; ++j) bzr[j] = bi[j * 16 + ncol] + br[j * 16 + ncol];
#pragma unroll
  for (int j = 0; j < 2; ++j) {
    bih[j] = bi[64 + j * 16 + ncol];
    brh[j] = br[64 + j * 16 + ncol];
  }

  v8f hreg[2];
#pragma unroll
  for (int c = 0; c < 2; ++c)
#pragma unroll
    for (int v = 0; v < 8; ++v) {
      float hv = hbuf[(size_t)(rbase + mbase + v) * DDIM + c * 16 + ncol];
      hreg[c][v] = hv;
      htile[wave][mbase + v][c * 16 + ncol] = (_Float16)hv;
    }

  // Stage x rows (2 lanes per row).
  {
    const int row = rbase + ncol;
    if (kx == 32) {
      int sr = idx ? idx[row] : row;
      const float* src = xA + sr * DDIM + lhalf * 16;
      _Float16* dst = &xtile[wave][ncol][lhalf * 16];
#pragma unroll
      for (int k = 0; k < 16; ++k) dst[k] = (_Float16)src[k];
    } else {
      const float* src = (lhalf ? xB : xA) + row * DDIM;
      _Float16* dst = &xtile[wave][ncol][lhalf * 32];
#pragma unroll
      for (int k = 0; k < 32; ++k) dst[k] = (_Float16)src[k];
    }
  }
  __syncthreads();

  v16h ax0 = ldA(&xtile[wave][0][0], 64, lane, 0);
  v16h ah  = ldA(&htile[wave][0][0], 32, lane, 0);
  v16h ax1 = ax0;
  if (kx == 64) ax1 = ldA(&xtile[wave][0][0], 64, lane, 32);

  v8f zr[4], ihv[2], rhv[2];
#pragma unroll
  for (int j = 0; j < 4; ++j) {
    v8f a;
#pragma unroll
    for (int v = 0; v < 8; ++v) a[v] = bzr[j];
    a = wmma16(ax0, ldB(&wiT[0][0], 64, lane, j, 0), a);
    if (kx == 64) a = wmma16(ax1, ldB(&wiT[0][0], 64, lane, j, 32), a);
    a = wmma16(ah, ldB(&uiT[0][0], 32, lane, j, 0), a);
    zr[j] = a;
  }
#pragma unroll
  for (int j = 0; j < 2; ++j) {
    v8f a;
#pragma unroll
    for (int v = 0; v < 8; ++v) a[v] = bih[j];
    a = wmma16(ax0, ldB(&wiT[0][0], 64, lane, 4 + j, 0), a);
    if (kx == 64) a = wmma16(ax1, ldB(&wiT[0][0], 64, lane, 4 + j, 32), a);
    ihv[j] = a;
    v8f b;
#pragma unroll
    for (int v = 0; v < 8; ++v) b[v] = brh[j];
    b = wmma16(ah, ldB(&uiT[0][0], 32, lane, 4 + j, 0), b);
    rhv[j] = b;
  }

#pragma unroll
  for (int c = 0; c < 2; ++c)
#pragma unroll
    for (int v = 0; v < 8; ++v) {
      float z = sigmoidf_(zr[c][v]);
      float r = sigmoidf_(zr[2 + c][v]);
      float hh = tanhf_(ihv[c][v] + r * rhv[c][v]);
      hbuf[(size_t)(rbase + mbase + v) * DDIM + c * 16 + ncol] =
          z * hreg[c][v] + (1.0f - z) * hh;
    }
}

// ---------------------------------------------------------------------------
// Readout: per flow, 8 timesteps through 32->16->16->1 MLP + capacity terms.
// ---------------------------------------------------------------------------
__global__ __launch_bounds__(256) void readout_kernel(
    const float* __restrict__ fss, const int* __restrict__ l2p,
    const float* __restrict__ link_cap, const float* __restrict__ avg_pkt, int s,
    const float* __restrict__ w1, const float* __restrict__ b1,
    const float* __restrict__ w2, const float* __restrict__ b2,
    const float* __restrict__ w3, const float* __restrict__ b3,
    float* __restrict__ out) {
  int f = blockIdx.x * 256 + threadIdx.x;
  if (f >= NFLOW) return;
  float acc = 0.0f, invs = 0.0f;
  for (int t = 0; t < LPATH; ++t) {
    const float* v = fss + (size_t)f * (LPATH + 1) * DDIM + (t + 1) * DDIM;
    float h1[16];
    for (int j = 0; j < 16; ++j) {
      float a = b1[j];
      for (int d = 0; d < DDIM; ++d) a += v[d] * w1[d * 16 + j];
      h1[j] = reluf_(a);
    }
    float h2[16];
    for (int j = 0; j < 16; ++j) {
      float a = b2[j];
      for (int i = 0; i < 16; ++i) a += h1[i] * w2[i * 16 + j];
      h2[j] = reluf_(a);
    }
    float ro = b3[0];
    for (int i = 0; i < 16; ++i) ro += h2[i] * w3[i];
    int li = l2p[f * LPATH + t];
    float cg = link_cap[li] * 1.0e9f;
    acc += ro / cg;
    invs += 1.0f / cg;
  }
  out[f] = acc + avg_pkt[f * SEGS + s] * invs;
}

// ---------------------------------------------------------------------------
extern "C" void kernel_launch(void* const* d_in, const int* in_sizes, int n_in,
                              void* d_out, int out_size, void* d_ws, size_t ws_size,
                              hipStream_t stream) {
  (void)in_sizes; (void)n_in; (void)out_size; (void)ws_size;
  const float* traffic  = (const float*)d_in[0];
  const float* pkt      = (const float*)d_in[1];
  const float* avg_pkt  = (const float*)d_in[2];
  const float* link_cap = (const float*)d_in[3];
  const float* hdr      = (const float*)d_in[4];
  // params dict leaves in insertion order
  const float* fe_w1 = (const float*)d_in[5];
  const float* fe_b1 = (const float*)d_in[6];
  const float* fe_w2 = (const float*)d_in[7];
  const float* fe_b2 = (const float*)d_in[8];
  const float* le_w1 = (const float*)d_in[9];
  const float* le_b1 = (const float*)d_in[10];
  const float* le_w2 = (const float*)d_in[11];
  const float* le_b2 = (const float*)d_in[12];
  const float* qe_w1 = (const float*)d_in[13];
  const float* qe_b1 = (const float*)d_in[14];
  const float* qe_w2 = (const float*)d_in[15];
  const float* qe_b2 = (const float*)d_in[16];
  const float* fg_wi = (const float*)d_in[17];
  const float* fg_ui = (const float*)d_in[18];
  const float* fg_bi = (const float*)d_in[19];
  const float* fg_br = (const float*)d_in[20];
  const float* qg_wi = (const float*)d_in[21];
  const float* qg_ui = (const float*)d_in[22];
  const float* qg_bi = (const float*)d_in[23];
  const float* qg_br = (const float*)d_in[24];
  const float* lg_wi = (const float*)d_in[25];
  const float* lg_ui = (const float*)d_in[26];
  const float* lg_bi = (const float*)d_in[27];
  const float* lg_br = (const float*)d_in[28];
  const float* wg_wi = (const float*)d_in[29];
  const float* wg_ui = (const float*)d_in[30];
  const float* wg_bi = (const float*)d_in[31];
  const float* wg_br = (const float*)d_in[32];
  const float* ro_w1 = (const float*)d_in[33];
  const float* ro_b1 = (const float*)d_in[34];
  const float* ro_w2 = (const float*)d_in[35];
  const float* ro_b2 = (const float*)d_in[36];
  const float* ro_w3 = (const float*)d_in[37];
  const float* ro_b3 = (const float*)d_in[38];
  const unsigned char* has = (const unsigned char*)d_in[39];
  const int* l2p   = (const int*)d_in[40];
  const int* p2l   = (const int*)d_in[41];
  const int* q2l   = (const int*)d_in[42];
  const int* btype = (const int*)d_in[43];

  float* ws = (float*)d_ws;
  float* init_flow   = ws;                                  // S*NF*D
  float* init_link   = init_flow + (size_t)SEGS * NFLOW * DDIM;   // S*NL*D
  float* init_queue  = init_link + (size_t)SEGS * NLINK * DDIM;   // NL*D
  float* queue_state = init_queue + (size_t)NLINK * DDIM;
  float* prev_queue  = queue_state + (size_t)NLINK * DDIM;
  float* link_state  = prev_queue + (size_t)NLINK * DDIM;
  float* flow_sum    = link_state + (size_t)NLINK * DDIM;
  float* fss         = flow_sum + (size_t)NLINK * DDIM;     // NF*(L+1)*D

  flow_embed_kernel<<<(NFLOW * SEGS + 255) / 256, 256, 0, stream>>>(
      traffic, pkt, has, fe_w1, fe_b1, fe_w2, fe_b2, init_flow);
  link_embed_kernel<<<NLINK / 256, 256, 0, stream>>>(
      traffic, pkt, p2l, link_cap, hdr, le_w1, le_b1, le_w2, le_b2, init_link);
  queue_embed_kernel<<<NLINK / 256, 256, 0, stream>>>(
      btype, qe_w1, qe_b1, qe_w2, qe_b2, init_queue, queue_state);

  const size_t nlbytes = (size_t)NLINK * DDIM * sizeof(float);
  for (int s = 0; s < SEGS; ++s) {
    hipMemcpyAsync(link_state, init_link + (size_t)s * NLINK * DDIM, nlbytes,
                   hipMemcpyDeviceToDevice, stream);
    hipMemcpyAsync(prev_queue, queue_state, nlbytes, hipMemcpyDeviceToDevice, stream);
    for (int it = 0; it < ITERS; ++it) {
      const float* hsrc = it ? (fss + LPATH * DDIM) : (init_flow + (size_t)s * NFLOW * DDIM);
      const int hstride = it ? (LPATH + 1) * DDIM : DDIM;
      flow_gru_kernel<<<NFLOW / 128, 256, 0, stream>>>(
          fg_wi, fg_ui, fg_bi, fg_br, l2p, queue_state, link_state, hsrc, hstride, fss);
      flow_sum_kernel<<<NLINK, 32, 0, stream>>>(fss, p2l, flow_sum);
      small_gru_kernel<<<NLINK / 128, 256, 0, stream>>>(
          qg_wi, qg_ui, qg_bi, qg_br, flow_sum, nullptr, nullptr, queue_state, 32);
      small_gru_kernel<<<NLINK / 128, 256, 0, stream>>>(
          lg_wi, lg_ui, lg_bi, lg_br, queue_state, nullptr, q2l, link_state, 32);
    }
    small_gru_kernel<<<NLINK / 128, 256, 0, stream>>>(
        wg_wi, wg_ui, wg_bi, wg_br, init_queue, prev_queue, nullptr, queue_state, 64);
    readout_kernel<<<NFLOW / 256, 256, 0, stream>>>(
        fss, l2p, link_cap, avg_pkt, s, ro_w1, ro_b1, ro_w2, ro_b2, ro_w3, ro_b3,
        (float*)d_out + (size_t)s * NFLOW);
  }
}
